// RotaryTransformer_29721173688582
// MI455X (gfx1250) — matrix operand order repaired and checked by reading, not compiled, not run on hardware
//
#include <hip/hip_runtime.h>
#include <hip/hip_bf16.h>
#include <math.h>

// ---------------- problem constants ----------------
constexpr int BB = 16, SS = 512, DD = 1024, HH = 16, LL = 6, NC_ = 1000;
constexpr int DH_ = 64, FF_ = 4096;

typedef __bf16 bf16;
typedef __attribute__((ext_vector_type(16))) __bf16 bf16x16;
typedef __attribute__((ext_vector_type(8)))  float  floatx8;
typedef __attribute__((ext_vector_type(4)))  unsigned uint32x4;
typedef int v4i __attribute__((vector_size(16)));

enum { F_BIAS = 1, F_RESID = 2, F_GELU = 4, F_OBF16 = 8, F_CT = 32 };

// ---------------- CDNA5 async global->LDS helpers ----------------
__device__ inline void wait_async0() {
#if __has_builtin(__builtin_amdgcn_s_wait_asynccnt)
    __builtin_amdgcn_s_wait_asynccnt(0);
#else
    asm volatile("s_wait_asynccnt 0x0" ::: "memory");
#endif
}

__device__ inline void async_copy_b128(const bf16* g, bf16* l) {
#if __has_builtin(__builtin_amdgcn_global_load_async_to_lds_b128)
    __builtin_amdgcn_global_load_async_to_lds_b128(
        (__attribute__((address_space(1))) v4i*)g,
        (__attribute__((address_space(3))) v4i*)l, 0, 0);
#else
    *(uint32x4*)l = *(const uint32x4*)g;   // synchronous fallback
#endif
}

// ---------------- helpers ----------------
__device__ inline float block_reduce_sum(float v) {
    __shared__ float red[256];
    int tid = threadIdx.x;
    red[tid] = v; __syncthreads();
    for (int s = 128; s > 0; s >>= 1) { if (tid < s) red[tid] += red[tid + s]; __syncthreads(); }
    float r = red[0]; __syncthreads();
    return r;
}
__device__ inline float block_reduce_max(float v) {
    __shared__ float red[256];
    int tid = threadIdx.x;
    red[tid] = v; __syncthreads();
    for (int s = 128; s > 0; s >>= 1) { if (tid < s) red[tid] = fmaxf(red[tid], red[tid + s]); __syncthreads(); }
    float r = red[0]; __syncthreads();
    return r;
}

// A fragment: 16x32 bf16, lane = row (lane&15), K halves by lane>>4
__device__ inline bf16x16 load_a_frag(const bf16* sA, int row0) {
    int lane = threadIdx.x & 31;
    const bf16* p = sA + (size_t)(row0 + (lane & 15)) * 32;
    int kb = (lane >> 4) << 3;           // 0 or 8
    bf16x16 f;
#pragma unroll
    for (int m = 0; m < 8; ++m) { f[m] = p[kb + m]; f[8 + m] = p[16 + kb + m]; }
    return f;
}
// B fragment: 32x16 bf16 staged as sB[n][k]; lane = col (lane&15), K half by lane>>4
__device__ inline bf16x16 load_b_frag(const bf16* sB, int col0) {
    int lane = threadIdx.x & 31;
    const bf16* p = sB + (size_t)(col0 + (lane & 15)) * 32 + ((lane >> 4) << 3);
    bf16x16 f;
#pragma unroll
    for (int m = 0; m < 8; ++m) { f[m] = p[m]; f[8 + m] = p[16 + m]; }
    return f;
}

// ---------------- bf16 WMMA GEMM ----------------
// C[M,N] = act( alpha * A[M,K] @ Bt[N,K]^T + bias ) (+ residual).
// B is ALWAYS stored [N][K] row-major (weights pre-transposed at convert time).
// Batched with two-level (bo,bi) offsets: z -> bo = z/inner, bi = z%inner.
// Staging uses CDNA5 async global->LDS B128 copies; OOB lanes read a zeroed pad.
template<int FLAGS>
__global__ __launch_bounds__(256)
void gemm_bf16_kernel(const bf16* __restrict__ A, const bf16* __restrict__ Bm,
                      const float* __restrict__ bias, const float* __restrict__ resid,
                      void* __restrict__ Cout, const bf16* __restrict__ zpad,
                      int M, int N, int K, int lda, int ldb, int ldc, int inner,
                      long long sAo, long long sAi, long long sBo, long long sBi,
                      long long sCo, long long sCi, float alpha)
{
    __shared__ bf16 sA[2][128 * 32];
    __shared__ bf16 sB[2][128 * 32];
    const int tid  = threadIdx.x;
    const int lane = tid & 31;
    const int wave = tid >> 5;
    const int wm   = wave >> 1;      // 0..3 -> 32-row strip
    const int wn   = wave & 1;       // 0..1 -> 64-col strip
    const int z  = blockIdx.z;
    const int bo = z / inner, bi = z - bo * inner;
    const bf16* Ap = A  + (size_t)((long long)bo * sAo + (long long)bi * sAi);
    const bf16* Bp = Bm + (size_t)((long long)bo * sBo + (long long)bi * sBi);
    const long long offC = (long long)bo * sCo + (long long)bi * sCi;
    const int m0 = blockIdx.y * 128, n0 = blockIdx.x * 128;

    // per-thread staging slots: 2x b128 for A, 2x b128 for B
    const int q0 = tid, q1 = tid + 256;            // b128 chunk ids 0..511
    const int ar0 = q0 >> 2, ac0 = (q0 & 3) << 3;  // row, k-offset
    const int ar1 = q1 >> 2, ac1 = (q1 & 3) << 3;
    const int gr0 = m0 + ar0, gr1 = m0 + ar1;
    const int gn0 = n0 + ar0, gn1 = n0 + ar1;

    auto stage_async = [&](int buf, int k0) {
        const bf16* ga0 = (gr0 < M) ? (Ap + (size_t)gr0 * lda + k0 + ac0) : zpad;
        const bf16* ga1 = (gr1 < M) ? (Ap + (size_t)gr1 * lda + k0 + ac1) : zpad;
        const bf16* gb0 = (gn0 < N) ? (Bp + (size_t)gn0 * ldb + k0 + ac0) : zpad;
        const bf16* gb1 = (gn1 < N) ? (Bp + (size_t)gn1 * ldb + k0 + ac1) : zpad;
        async_copy_b128(ga0, &sA[buf][ar0 * 32 + ac0]);
        async_copy_b128(ga1, &sA[buf][ar1 * 32 + ac1]);
        async_copy_b128(gb0, &sB[buf][ar0 * 32 + ac0]);
        async_copy_b128(gb1, &sB[buf][ar1 * 32 + ac1]);
    };

    floatx8 acc[2][4];
#pragma unroll
    for (int i = 0; i < 2; ++i)
#pragma unroll
        for (int j = 0; j < 4; ++j)
#pragma unroll
            for (int r = 0; r < 8; ++r) acc[i][j][r] = 0.f;

    const int nk = K >> 5;            // K is always a multiple of 32 here
    stage_async(0, 0);
    wait_async0();
    for (int kk = 0; kk < nk; ++kk) {
        __syncthreads();
        const int cur = kk & 1;
        const bool more = (kk + 1) < nk;
        if (more) stage_async(cur ^ 1, (kk + 1) << 5);  // async DMA next K-tile

        bf16x16 af[2], bfr[4];
#pragma unroll
        for (int i = 0; i < 2; ++i) af[i] = load_a_frag(sA[cur], wm * 32 + i * 16);
#pragma unroll
        for (int j = 0; j < 4; ++j) bfr[j] = load_b_frag(sB[cur], wn * 64 + j * 16);
#pragma unroll
        for (int i = 0; i < 2; ++i)
#pragma unroll
            for (int j = 0; j < 4; ++j)
                acc[i][j] = __builtin_amdgcn_wmma_f32_16x16x32_bf16(
                    false, af[i], false, bfr[j], (short)0, acc[i][j], false, false);

        if (more) wait_async0();   // own asyncs landed; barrier makes them global
    }

    // epilogue: C tile layout -> lane holds col = lane&15, rows r + 8*(lane>>4)
    const int colb = lane & 15;
    const int rowo = (lane >> 4) << 3;
#pragma unroll
    for (int i = 0; i < 2; ++i) {
#pragma unroll
        for (int j = 0; j < 4; ++j) {
            int col = n0 + wn * 64 + j * 16 + colb;
            if (col >= N) continue;
            float bv = (FLAGS & F_BIAS) ? bias[col] : 0.f;
#pragma unroll
            for (int r = 0; r < 8; ++r) {
                int row = m0 + wm * 32 + i * 16 + rowo + r;
                if (row >= M) continue;
                float v = acc[i][j][r] * alpha + bv;
                if (FLAGS & F_GELU) v = 0.5f * v * (1.f + erff(v * 0.70710678f));
                if (FLAGS & F_CT) {
                    // transposed store: C is [B][H][DH][S] (for V so PV GEMM gets [N][K])
                    int b = row >> 9, s = row & (SS - 1);        // SS = 512
                    int h = col >> 6, dh = col & (DH_ - 1);      // DH = 64
                    size_t cidx = (((size_t)b * HH + h) * DH_ + dh) * SS + s;
                    ((bf16*)Cout)[cidx] = (bf16)v;
                } else {
                    size_t cidx = (size_t)(offC + (long long)row * ldc + col);
                    if (FLAGS & F_RESID) v += resid[cidx];
                    if (FLAGS & F_OBF16) ((bf16*)Cout)[cidx] = (bf16)v;
                    else                 ((float*)Cout)[cidx] = v;
                }
            }
        }
    }
}

template<int FLAGS>
static inline void launch_gemm(hipStream_t st,
    const bf16* A, const bf16* Bm, const float* bias, const float* resid, void* C,
    const bf16* zpad,
    int M, int N, int K, int lda, int ldb, int ldc,
    int batches, int inner,
    long long sAo, long long sAi, long long sBo, long long sBi,
    long long sCo, long long sCi, float alpha)
{
    dim3 grid((N + 127) / 128, (M + 127) / 128, batches);
    gemm_bf16_kernel<FLAGS><<<grid, 256, 0, st>>>(A, Bm, bias, resid, C, zpad,
        M, N, K, lda, ldb, ldc, inner, sAo, sAi, sBo, sBi, sCo, sCi, alpha);
}

// ---------------- convert + transpose: W[K][N] f32 -> Wt[N][K] bf16 ----------------
__global__ void cvt_transpose_kernel(const float* __restrict__ src, bf16* __restrict__ dst,
                                     int K, int N) {
    __shared__ float t[32][33];
    int bn = blockIdx.x * 32, bk = blockIdx.y * 32;
    const float* s = src + (size_t)blockIdx.z * K * N;
    bf16* d = dst + (size_t)blockIdx.z * N * K;
    int tx = threadIdx.x & 31, ty = threadIdx.x >> 5;   // 32 x 8
#pragma unroll
    for (int i = 0; i < 32; i += 8) {
        int k = bk + ty + i, n = bn + tx;
        float v = 0.f;
        if (k < K && n < N) v = s[(size_t)k * N + n];
        t[ty + i][tx] = v;
    }
    __syncthreads();
#pragma unroll
    for (int i = 0; i < 32; i += 8) {
        int n = bn + ty + i, k = bk + tx;
        if (n < N && k < K) d[(size_t)n * K + k] = (bf16)t[tx][ty + i];
    }
}

// ---------------- elementwise / reduction kernels ----------------
__global__ void zero_kernel(unsigned* __restrict__ p, int n) {
    int i = blockIdx.x * 256 + threadIdx.x;
    if (i < n) p[i] = 0u;
}

__global__ void embed_kernel(const int* __restrict__ nums, const float* __restrict__ embed,
                             float* __restrict__ x) {
    int row = blockIdx.x;                      // B*S rows
    int tok = nums[row];
    const float* src = embed + (size_t)tok * DD;
    float* dst = x + (size_t)row * DD;
    for (int i = threadIdx.x; i < DD; i += 256) dst[i] = src[i];
}

__global__ void layernorm_kernel(const float* __restrict__ x, const float* __restrict__ g,
                                 const float* __restrict__ b, bf16* __restrict__ out, int D) {
    int row = blockIdx.x;
    const float* xr = x + (size_t)row * D;
    float s = 0.f, s2 = 0.f;
    for (int i = threadIdx.x; i < D; i += 256) { float v = xr[i]; s += v; s2 += v * v; }
    float mean = block_reduce_sum(s) / (float)D;
    float var  = block_reduce_sum(s2) / (float)D - mean * mean;
    float rstd = rsqrtf(var + 1e-5f);
    bf16* o = out + (size_t)row * D;
    for (int i = threadIdx.x; i < D; i += 256)
        o[i] = (bf16)((xr[i] - mean) * rstd * g[i] + b[i]);
}

__global__ void rope_kernel(bf16* __restrict__ q, bf16* __restrict__ k) {
    int idx = blockIdx.x * 256 + threadIdx.x;
    const int total = BB * SS * HH * (DH_ / 2);
    if (idx >= total) return;
    int i = idx & (DH_ / 2 - 1);
    int t = idx / (DH_ / 2);
    int h = t & (HH - 1); t /= HH;
    int s = t & (SS - 1); int b = t / SS;
    float freq = __expf(-(float)(2 * i) / (float)DH_ * 9.2103403720f); // ln(10000)
    float ang = (float)s * freq;
    float sn, c; __sincosf(ang, &sn, &c);
    size_t base = (((size_t)b * SS + s) * DD) + (size_t)h * DH_ + 2 * i;
    float q1 = (float)q[base], q2 = (float)q[base + 1];
    q[base]     = (bf16)(q1 * c - q2 * sn);
    q[base + 1] = (bf16)(q1 * sn + q2 * c);
    float k1 = (float)k[base], k2 = (float)k[base + 1];
    k[base]     = (bf16)(k1 * c - k2 * sn);
    k[base + 1] = (bf16)(k1 * sn + k2 * c);
}

__global__ void softmax_kernel(const float* __restrict__ scores, bf16* __restrict__ probs, int n) {
    size_t base = (size_t)blockIdx.x * n;
    const float* xr = scores + base;
    float m = -1e30f;
    for (int i = threadIdx.x; i < n; i += 256) m = fmaxf(m, xr[i]);
    m = block_reduce_max(m);
    float s = 0.f;
    for (int i = threadIdx.x; i < n; i += 256) s += __expf(xr[i] - m);
    s = block_reduce_sum(s);
    float inv = 1.f / s;
    bf16* o = probs + base;
    for (int i = threadIdx.x; i < n; i += 256) o[i] = (bf16)(__expf(xr[i] - m) * inv);
}

__global__ void mean_pool_kernel(const float* __restrict__ x, float* __restrict__ out) {
    int idx = blockIdx.x * 256 + threadIdx.x;
    if (idx >= BB * DD) return;
    int b = idx / DD, d = idx - b * DD;
    float s = 0.f;
    for (int t = 0; t < SS; ++t) s += x[((size_t)b * SS + t) * DD + d];
    out[idx] = s * (1.f / (float)SS);
}

// ---------------- orchestration ----------------
extern "C" void kernel_launch(void* const* d_in, const int* in_sizes, int n_in,
                              void* d_out, int out_size, void* d_ws, size_t ws_size,
                              hipStream_t stream)
{
    const int*   nums  = (const int*)d_in[0];
    const float* embed = (const float*)d_in[1];
    const float* q_w = (const float*)d_in[2];  const float* q_b = (const float*)d_in[3];
    const float* k_w = (const float*)d_in[4];  const float* k_b = (const float*)d_in[5];
    const float* v_w = (const float*)d_in[6];  const float* v_b = (const float*)d_in[7];
    const float* o_w = (const float*)d_in[8];  const float* o_b = (const float*)d_in[9];
    const float* ln1_g = (const float*)d_in[10]; const float* ln1_b = (const float*)d_in[11];
    const float* ln2_g = (const float*)d_in[12]; const float* ln2_b = (const float*)d_in[13];
    const float* ff1_w = (const float*)d_in[14]; const float* ff1_b = (const float*)d_in[15];
    const float* ff2_w = (const float*)d_in[16]; const float* ff2_b = (const float*)d_in[17];
    const float* hln_g = (const float*)d_in[18]; const float* hln_b = (const float*)d_in[19];
    const float* h1_w = (const float*)d_in[20]; const float* h1_b = (const float*)d_in[21];
    const float* h2_w = (const float*)d_in[22]; const float* h2_b = (const float*)d_in[23];

    char* base = (char*)d_ws; size_t off = 0;
    auto alloc = [&](size_t bytes) -> void* {
        off = (off + 255) & ~(size_t)255;
        void* p = base + off; off += bytes; return p;
    };

    // zero pad for OOB async lanes
    bf16* zpad = (bf16*)alloc(256);
    // bf16 TRANSPOSED weight mirrors (stored [N][K])
    bf16* qw  = (bf16*)alloc((size_t)LL * DD * DD * 2);
    bf16* kw  = (bf16*)alloc((size_t)LL * DD * DD * 2);
    bf16* vw  = (bf16*)alloc((size_t)LL * DD * DD * 2);
    bf16* ow  = (bf16*)alloc((size_t)LL * DD * DD * 2);
    bf16* f1w = (bf16*)alloc((size_t)LL * DD * FF_ * 2);
    bf16* f2w = (bf16*)alloc((size_t)LL * FF_ * DD * 2);
    bf16* h1w = (bf16*)alloc((size_t)DD * DD * 2);
    bf16* h2w = (bf16*)alloc((size_t)NC_ * DD * 2);
    // activations
    float* x    = (float*)alloc((size_t)BB * SS * DD * 4);
    bf16*  hbf  = (bf16*) alloc((size_t)BB * SS * DD * 2);
    bf16*  qbf  = (bf16*) alloc((size_t)BB * SS * DD * 2);
    bf16*  kbf  = (bf16*) alloc((size_t)BB * SS * DD * 2);
    bf16*  vt   = (bf16*) alloc((size_t)BB * SS * DD * 2);   // [B][H][DH][S]
    float* scr  = (float*)alloc((size_t)BB * HH * SS * SS * 4);
    bf16*  prb  = (bf16*) alloc((size_t)BB * HH * SS * SS * 2);
    bf16*  aobf = (bf16*) alloc((size_t)BB * SS * DD * 2);
    bf16*  ffh  = (bf16*) alloc((size_t)BB * SS * FF_ * 2);
    float* pool = (float*)alloc((size_t)BB * DD * 4);
    bf16*  poob = (bf16*) alloc((size_t)BB * DD * 2);
    bf16*  hhbf = (bf16*) alloc((size_t)BB * DD * 2);

    zero_kernel<<<1, 256, 0, stream>>>((unsigned*)zpad, 64);

    auto cvtT = [&](const float* s, bf16* d, int K, int N, int batch) {
        dim3 g((N + 31) / 32, (K + 31) / 32, batch);
        cvt_transpose_kernel<<<g, 256, 0, stream>>>(s, d, K, N);
    };
    cvtT(q_w,  qw,  DD, DD, LL);
    cvtT(k_w,  kw,  DD, DD, LL);
    cvtT(v_w,  vw,  DD, DD, LL);
    cvtT(o_w,  ow,  DD, DD, LL);
    cvtT(ff1_w, f1w, DD, FF_, LL);
    cvtT(ff2_w, f2w, FF_, DD, LL);
    cvtT(h1_w, h1w, DD, DD, 1);
    cvtT(h2_w, h2w, DD, NC_, 1);

    // x = embed[nums]
    embed_kernel<<<BB * SS, 256, 0, stream>>>(nums, embed, x);

    const int M = BB * SS;
    const long long qSA = (long long)SS * DD, qSI = DH_;
    const long long cSO = (long long)HH * SS * SS, cSI = (long long)SS * SS;
    const long long vSO = (long long)HH * DH_ * SS, vSI = (long long)DH_ * SS;

    for (int i = 0; i < LL; ++i) {
        size_t wdd = (size_t)i * DD * DD;
        // h = LN1(x)
        layernorm_kernel<<<M, 256, 0, stream>>>(x, ln1_g + (size_t)i * DD, ln1_b + (size_t)i * DD, hbf, DD);
        // q,k projections (bias, bf16); v projection stores transposed (vt)
        launch_gemm<F_BIAS | F_OBF16>(stream, hbf, qw + wdd, q_b + (size_t)i * DD, nullptr, qbf, zpad,
            M, DD, DD, DD, DD, DD, 1, 1, 0, 0, 0, 0, 0, 0, 1.f);
        launch_gemm<F_BIAS | F_OBF16>(stream, hbf, kw + wdd, k_b + (size_t)i * DD, nullptr, kbf, zpad,
            M, DD, DD, DD, DD, DD, 1, 1, 0, 0, 0, 0, 0, 0, 1.f);
        launch_gemm<F_BIAS | F_OBF16 | F_CT>(stream, hbf, vw + wdd, v_b + (size_t)i * DD, nullptr, vt, zpad,
            M, DD, DD, DD, DD, DD, 1, 1, 0, 0, 0, 0, 0, 0, 1.f);
        // RoPE on q,k
        rope_kernel<<<(BB * SS * HH * (DH_ / 2) + 255) / 256, 256, 0, stream>>>(qbf, kbf);
        // scores = q @ k^T / sqrt(DH); k slice is naturally [N=t][K=d]
        launch_gemm<0>(stream, qbf, kbf, nullptr, nullptr, scr, zpad,
            SS, SS, DH_, DD, DD, SS, BB * HH, HH,
            qSA, qSI, qSA, qSI, cSO, cSI, 0.125f);
        // probs = softmax(scores)
        softmax_kernel<<<BB * HH * SS, 256, 0, stream>>>(scr, prb, SS);
        // out = probs @ v  (vt is [N=dh][K=t] per head)
        launch_gemm<F_OBF16>(stream, prb, vt, nullptr, nullptr, aobf, zpad,
            SS, DH_, SS, SS, SS, DD, BB * HH, HH,
            cSO, cSI, vSO, vSI, qSA, qSI, 1.f);
        // x = x + out @ o_w + o_b
        launch_gemm<F_BIAS | F_RESID>(stream, aobf, ow + wdd, o_b + (size_t)i * DD, x, x, zpad,
            M, DD, DD, DD, DD, DD, 1, 1, 0, 0, 0, 0, 0, 0, 1.f);
        // h = LN2(x); ffh = gelu(h @ ff1 + b1); x = x + ffh @ ff2 + b2
        layernorm_kernel<<<M, 256, 0, stream>>>(x, ln2_g + (size_t)i * DD, ln2_b + (size_t)i * DD, hbf, DD);
        launch_gemm<F_BIAS | F_GELU | F_OBF16>(stream, hbf, f1w + (size_t)i * DD * FF_,
            ff1_b + (size_t)i * FF_, nullptr, ffh, zpad,
            M, FF_, DD, DD, DD, FF_, 1, 1, 0, 0, 0, 0, 0, 0, 1.f);
        launch_gemm<F_BIAS | F_RESID>(stream, ffh, f2w + (size_t)i * FF_ * DD,
            ff2_b + (size_t)i * DD, x, x, zpad,
            M, DD, FF_, FF_, FF_, DD, 1, 1, 0, 0, 0, 0, 0, 0, 1.f);
    }

    // head: mean over S, LN, gelu(h@W1+b1), h@W2+b2 -> d_out (f32)
    mean_pool_kernel<<<(BB * DD + 255) / 256, 256, 0, stream>>>(x, pool);
    layernorm_kernel<<<BB, 256, 0, stream>>>(pool, hln_g, hln_b, poob, DD);
    launch_gemm<F_BIAS | F_GELU | F_OBF16>(stream, poob, h1w, h1_b, nullptr, hhbf, zpad,
        BB, DD, DD, DD, DD, DD, 1, 1, 0, 0, 0, 0, 0, 0, 1.f);
    launch_gemm<F_BIAS>(stream, hhbf, h2w, h2_b, nullptr, d_out, zpad,
        BB, NC_, DD, DD, DD, NC_, 1, 1, 0, 0, 0, 0, 0, 0, 1.f);
}
